// NeuralGraph_67087389163569
// MI455X (gfx1250) — matrix operand order repaired and checked by reading, not compiled, not run on hardware
//
#include <hip/hip_runtime.h>
#include <hip/hip_bf16.h>

typedef __attribute__((ext_vector_type(16))) _Float16 v16h;
typedef __attribute__((ext_vector_type(8)))  _Float16 v8h;
typedef __attribute__((ext_vector_type(8)))  float    v8f;

#define MAXV 100.0f

__device__ __forceinline__ float clipf(float v) {
    return fminf(fmaxf(v, -MAXV), MAXV);
}

// ---------------------------------------------------------------------------
// Workspace zeroing (graph-replay safe: re-zero every launch)
// ---------------------------------------------------------------------------
__global__ __launch_bounds__(256) void zero_kernel(float* __restrict__ p, long n) {
    long i = (long)blockIdx.x * blockDim.x + threadIdx.x;
    long stride = (long)gridDim.x * blockDim.x;
    for (; i < n; i += stride) p[i] = 0.0f;
}

// ---------------------------------------------------------------------------
// Degree computation with run-length merging (sources come in runs of 1040)
// ---------------------------------------------------------------------------
__global__ __launch_bounds__(256) void deg_kernel(const int* __restrict__ src,
                                                  const int* __restrict__ tgt,
                                                  float* __restrict__ outdeg,
                                                  float* __restrict__ indeg, long E) {
    long i0 = ((long)blockIdx.x * blockDim.x + threadIdx.x) * 32;
    if (i0 >= E) return;
    long iend = i0 + 32; if (iend > E) iend = E;

    int cur = src[i0]; float cnt = 1.0f;
    for (long i = i0 + 1; i < iend; ++i) {
        int s = src[i];
        if (s == cur) { cnt += 1.0f; }
        else { atomicAdd(&outdeg[cur], cnt); cur = s; cnt = 1.0f; }
    }
    atomicAdd(&outdeg[cur], cnt);

    cur = tgt[i0]; cnt = 1.0f;
    for (long i = i0 + 1; i < iend; ++i) {
        int t = tgt[i];
        if (t == cur) { cnt += 1.0f; }
        else { atomicAdd(&indeg[cur], cnt); cur = t; cnt = 1.0f; }
    }
    atomicAdd(&indeg[cur], cnt);
}

// ---------------------------------------------------------------------------
// Message MLP over edges via v_wmma_f32_16x16x32_f16. Persistent waves:
// each wave loops over many 16-edge tiles, amortizing the wave-invariant
// weight-fragment preload. (b, e) maintained incrementally (no per-tile i64 div).
//   Layer 1: A(16x64 f16, features padded) x W1^T(64x32) -> 16x32 hidden (f32)
//   Layer 2: relu(hidden) re-laid-out via LDS -> A(16x32) x W2^T(32x32pad) -> 16x24
// ---------------------------------------------------------------------------
__global__ __launch_bounds__(256) void msg_kernel(
    const float* __restrict__ nodes, const float* __restrict__ edges,
    const float* __restrict__ w1, const float* __restrict__ b1,
    const float* __restrict__ w2, const float* __restrict__ b2,
    const int* __restrict__ src, const int* __restrict__ tgt,
    const float* __restrict__ outdeg, const float* __restrict__ indeg,
    float* __restrict__ agg_a, float* __restrict__ agg_b,
    float* __restrict__ out_edges,
    int N, int Bn, long E, long BE, long numTiles)
{
    // Per-wave staging (8 waves / block)
    __shared__ __align__(16) _Float16 hstage[8][16 * 32]; // hidden tile, row-major
    __shared__ __align__(16) float    mstage[8][16 * 8];  // m_ab tile for coalesced stores
    __shared__ int idxA[8][16];                           // b*N + src per row
    __shared__ int idxB[8][16];                           // b*N + tgt per row

    const int lane = threadIdx.x & 31;
    const int wid  = threadIdx.x >> 5;
    const int half = lane >> 4;   // K-half selector for A frags / M-half for C/D
    const int col  = lane & 15;

    // ---- Load wave-invariant B fragments + bias C fragments (registers) ----
    v16h bw1[2][2];   // [N-tile][K-tile]
    v16h bw2[2];      // [N-tile]
    v8f  c1[2], c2[2];
#pragma unroll
    for (int nt = 0; nt < 2; ++nt) {
        const int hcol = col + 16 * nt;            // hidden index 0..31
#pragma unroll
        for (int i = 0; i < 16; ++i) {
            const int K0 = (i < 8) ? (8 * half + i) : (16 + 8 * half + (i - 8)); // K-tile 0: table order, K in 0..31
            bw1[nt][0][i] = (_Float16)w1[hcol * 34 + K0];
            const int K1 = 32 + 16 * half + i;     // K-tile 1: only K=32,33 valid
            const float wv1 = w1[hcol * 34 + ((K1 < 34) ? K1 : 0)];
            bw1[nt][1][i] = (_Float16)((K1 < 34) ? wv1 : 0.0f);
        }
        const float bias1 = b1[hcol];
#pragma unroll
        for (int r = 0; r < 8; ++r) c1[nt][r] = bias1;

        const int ocol = col + 16 * nt;            // output index, valid if < 24
#pragma unroll
        for (int i = 0; i < 16; ++i) {
            const float wv2 = w2[((ocol < 24) ? ocol : 0) * 32 + ((i < 8) ? (8 * half + i) : (16 + 8 * half + (i - 8)))];
            bw2[nt][i] = (_Float16)((ocol < 24) ? wv2 : 0.0f);
        }
        const float bias2 = (ocol < 24) ? b2[ocol] : 0.0f;
#pragma unroll
        for (int r = 0; r < 8; ++r) c2[nt][r] = bias2;
    }

    const long wgid   = (long)blockIdx.x * 8 + wid;
    const long nwaves = (long)gridDim.x * 8;
    const long gstep  = nwaves * 16;               // per-lane edge-id stride per iter

    if (wgid >= numTiles) return;

    // One-time (b, e) split; maintained incrementally afterwards.
    long g  = wgid * 16 + col;
    {
        // nothing else
    }
    long gi = (g < BE) ? g : (BE - 1);
    int  bL = (int)(gi / E);
    long eL = gi - (long)bL * E;

    for (long tile = wgid; tile < numTiles; tile += nwaves) {
        const long g0 = tile * 16;

        // ---- Gather A fragments directly in WMMA layout (no LDS needed) ----
        // Lane (half, col): row M = col. half0 holds K 0..7,16..23; half1 K 8..15,24..31.
        // Features: [0:12]=nodes[src], [12:24]=nodes[tgt], [24:32]=edge, 32=od, 33=idg.
        int  bc = bL;
        long ec = eL;
        long gc = g0 + col;
        if (gc >= BE) { bc = Bn - 1; ec = E - 1; gc = BE - 1; }  // tail clamp (copies only)
        const int  s  = src[ec];
        const int  t  = tgt[ec];
        const float* ns  = nodes + ((long)bc * N + s) * 12;
        const float* ntp = nodes + ((long)bc * N + t) * 12;
        const float* ed  = edges + gc * 8;

        v16h a0, a1;
#pragma unroll
        for (int i = 0; i < 16; ++i) a1[i] = (_Float16)0.0f;
        if (half == 0) {
#pragma unroll
            for (int i = 0; i < 8; ++i) a0[i]     = (_Float16)ns[i];       // K 0..7
#pragma unroll
            for (int i = 0; i < 8; ++i) a0[8 + i] = (_Float16)ntp[4 + i];  // K 16..23
            a1[0] = (_Float16)outdeg[s];                                    // K 32
            a1[1] = (_Float16)indeg[t];                                     // K 33
            idxA[wid][col] = bc * N + s;
            idxB[wid][col] = bc * N + t;
        } else {
#pragma unroll
            for (int i = 0; i < 4; ++i) a0[i]     = (_Float16)ns[8 + i];   // K 8..11
#pragma unroll
            for (int i = 0; i < 4; ++i) a0[4 + i] = (_Float16)ntp[i];      // K 12..15
#pragma unroll
            for (int i = 0; i < 8; ++i) a0[8 + i] = (_Float16)ed[i];       // K 24..31
        }

        // ---- Layer 1: two N-tiles, two K-tiles each ----
        v8f h0 = __builtin_amdgcn_wmma_f32_16x16x32_f16(false, a0, false, bw1[0][0], (short)0, c1[0], false, false);
        h0     = __builtin_amdgcn_wmma_f32_16x16x32_f16(false, a1, false, bw1[0][1], (short)0, h0,    false, false);
        v8f h1 = __builtin_amdgcn_wmma_f32_16x16x32_f16(false, a0, false, bw1[1][0], (short)0, c1[1], false, false);
        h1     = __builtin_amdgcn_wmma_f32_16x16x32_f16(false, a1, false, bw1[1][1], (short)0, h1,    false, false);

        // ---- ReLU + D-layout -> A-layout bounce through per-wave LDS ----
        {
            const int mb = half * 8;
#pragma unroll
            for (int r = 0; r < 8; ++r) {
                hstage[wid][(mb + r) * 32 + col]      = (_Float16)fmaxf(h0[r], 0.0f);
                hstage[wid][(mb + r) * 32 + col + 16] = (_Float16)fmaxf(h1[r], 0.0f);
            }
        }
        asm volatile("s_wait_dscnt 0" ::: "memory");   // wave-local LDS RAW fence

        v16h a2;
        {
            const char* hb = (const char*)&hstage[wid][0];
            const v8h lo = *(const v8h*)(hb + col * 64 + half * 16);        // K 8h..8h+7
            const v8h hi = *(const v8h*)(hb + col * 64 + 32 + half * 16);   // K 16+8h..
#pragma unroll
            for (int i = 0; i < 8; ++i) { a2[i] = lo[i]; a2[8 + i] = hi[i]; }
        }

        // ---- Layer 2: m[:,0:8]=m_a, m[:,8:16]=m_b, m[:,16:24]=m_ab ----
        v8f o0 = __builtin_amdgcn_wmma_f32_16x16x32_f16(false, a2, false, bw2[0], (short)0, c2[0], false, false);
        v8f o1 = __builtin_amdgcn_wmma_f32_16x16x32_f16(false, a2, false, bw2[1], (short)0, c2[1], false, false);

        // ---- m_ab: stage to LDS, then fully coalesced 512B read-modify-write ----
        if (col < 8) {
#pragma unroll
            for (int r = 0; r < 8; ++r)
                mstage[wid][(half * 8 + r) * 8 + col] = o1[r];
        }
        asm volatile("s_wait_dscnt 0" ::: "memory");
        {
            const long fo = g0 * 8 + lane * 4;
            if (fo < BE * 8) {
                const float* ep = edges + fo;
                const float* mm = &mstage[wid][lane * 4];
                float* op = out_edges + fo;
                op[0] = clipf(ep[0] + mm[0]);
                op[1] = clipf(ep[1] + mm[1]);
                op[2] = clipf(ep[2] + mm[2]);
                op[3] = clipf(ep[3] + mm[3]);
            }
        }

        // ---- Scatter m_a / m_b into aggregators ----
        int ia[8], ib[8];
        const int mb = half * 8;
#pragma unroll
        for (int r = 0; r < 8; ++r) { ia[r] = idxA[wid][mb + r]; ib[r] = idxB[wid][mb + r]; }
        const int ia0 = idxA[wid][0];
        bool uni = (g0 + 16) <= BE;
#pragma unroll
        for (int r = 0; r < 8; ++r) uni = uni && (ia[r] == ia0);
        if (__all(uni)) {
            // Uniform-source tile (dominant case): reduce 16 rows in registers,
            // one atomic per channel instead of 128.
            float sA = 0.0f;
#pragma unroll
            for (int r = 0; r < 8; ++r) sA += o0[r];
            sA += __shfl_xor(sA, 16);
            if (lane < 8) atomicAdd(agg_a + (long)ia0 * 8 + lane, sA);
        } else if (col < 8) {
#pragma unroll
            for (int r = 0; r < 8; ++r) {
                const long g2 = g0 + mb + r;
                if (g2 < BE) atomicAdd(agg_a + (long)ia[r] * 8 + col, o0[r]);
            }
        }
        if (col >= 8) {  // m_b: targets cycle, no uniformity to exploit
            const int ch = col - 8;
#pragma unroll
            for (int r = 0; r < 8; ++r) {
                const long g2 = g0 + mb + r;
                if (g2 < BE) atomicAdd(agg_b + (long)ib[r] * 8 + ch, o0[r]);
            }
        }

        // ---- Advance incremental (b, e) split for the next tile ----
        eL += gstep;
        while (eL >= E) { eL -= E; ++bL; }
    }
}

// ---------------------------------------------------------------------------
// Node update MLP (28 -> 16 -> 8): only B*N = ~2210 instances, plain VALU.
// ---------------------------------------------------------------------------
__global__ __launch_bounds__(256) void upd_kernel(
    const float* __restrict__ nodes,
    const float* __restrict__ outdeg, const float* __restrict__ indeg,
    const float* __restrict__ agg_a, const float* __restrict__ agg_b,
    const float* __restrict__ w1, const float* __restrict__ b1,
    const float* __restrict__ w2, const float* __restrict__ b2,
    float* __restrict__ out_nodes, long N, long BN)
{
    const long i = (long)blockIdx.x * blockDim.x + threadIdx.x;
    if (i >= BN) return;
    const int n = (int)(i % N);

    float x[28], nd[12];
    const float oda = fmaxf(outdeg[n], 1.0f);
    const float idb = fmaxf(indeg[n], 1.0f);
#pragma unroll
    for (int c = 0; c < 8; ++c)  x[c]     = agg_a[i * 8 + c] / oda;
#pragma unroll
    for (int c = 0; c < 8; ++c)  x[8 + c] = agg_b[i * 8 + c] / idb;
#pragma unroll
    for (int c = 0; c < 12; ++c) { nd[c] = nodes[i * 12 + c]; x[16 + c] = nd[c]; }

    float h[16];
#pragma unroll
    for (int j = 0; j < 16; ++j) {
        float acc = b1[j];
#pragma unroll
        for (int f = 0; f < 28; ++f) acc = fmaf(w1[j * 28 + f], x[f], acc);
        h[j] = fmaxf(acc, 0.0f);
    }
#pragma unroll
    for (int o = 0; o < 8; ++o) {
        float acc = b2[o];
#pragma unroll
        for (int j = 0; j < 16; ++j) acc = fmaf(w2[o * 16 + j], h[j], acc);
        out_nodes[i * 12 + o] = clipf(nd[o] + acc);
    }
#pragma unroll
    for (int c = 8; c < 12; ++c) out_nodes[i * 12 + c] = nd[c];
}

// ---------------------------------------------------------------------------
extern "C" void kernel_launch(void* const* d_in, const int* in_sizes, int n_in,
                              void* d_out, int out_size, void* d_ws, size_t ws_size,
                              hipStream_t stream) {
    const float* nodes  = (const float*)d_in[0];
    const float* edges  = (const float*)d_in[1];
    const float* msg_w1 = (const float*)d_in[2];
    const float* msg_b1 = (const float*)d_in[3];
    const float* msg_w2 = (const float*)d_in[4];
    const float* msg_b2 = (const float*)d_in[5];
    const float* upd_w1 = (const float*)d_in[6];
    const float* upd_b1 = (const float*)d_in[7];
    const float* upd_w2 = (const float*)d_in[8];
    const float* upd_b2 = (const float*)d_in[9];
    const int*   src    = (const int*)d_in[10];
    const int*   tgt    = (const int*)d_in[11];

    const long E  = in_sizes[10];
    const long B  = in_sizes[1] / (E * 8);
    const long N  = in_sizes[0] / (B * 12);
    const long BE = B * E;
    const long BN = B * N;

    // Workspace layout: outdeg[N] | indeg[N] | agg_a[B*N*8] | agg_b[B*N*8]
    float* ws     = (float*)d_ws;
    float* outdeg = ws;
    float* indeg  = ws + N;
    float* agg_a  = ws + 2 * N;
    float* agg_b  = agg_a + BN * 8;
    const long wsFloats = 2 * N + 2 * BN * 8;

    float* out_nodes = (float*)d_out;
    float* out_edges = out_nodes + BN * 12;

    {
        int blk = (int)((wsFloats + 255) / 256);
        if (blk > 1024) blk = 1024;
        zero_kernel<<<blk, 256, 0, stream>>>(ws, wsFloats);
    }
    {
        long chunks = (E + 31) / 32;
        int blk = (int)((chunks + 255) / 256);
        deg_kernel<<<blk, 256, 0, stream>>>(src, tgt, outdeg, indeg, E);
    }
    {
        const long numTiles = (BE + 15) / 16;
        int blk = (int)((numTiles + 7) / 8);    // 8 waves per block
        if (blk > 2048) blk = 2048;             // persistent waves: ~9 tiles/wave,
                                                // amortizes weight-fragment preload
        msg_kernel<<<blk, 256, 0, stream>>>(nodes, edges, msg_w1, msg_b1, msg_w2,
                                            msg_b2, src, tgt, outdeg, indeg,
                                            agg_a, agg_b, out_edges,
                                            (int)N, (int)B, E, BE, numTiles);
    }
    {
        int blk = (int)((BN + 255) / 256);
        upd_kernel<<<blk, 256, 0, stream>>>(nodes, outdeg, indeg, agg_a, agg_b,
                                            upd_w1, upd_b1, upd_w2, upd_b2,
                                            out_nodes, N, BN);
    }
}
